// _Memory_Block_7103875907886
// MI455X (gfx1250) — matrix operand order repaired and checked by reading, not compiled, not run on hardware
//
#include <hip/hip_runtime.h>
#include <math.h>

// ---- problem constants (from reference setup_inputs) ----
constexpr int B_  = 8;
constexpr int C_  = 256;
constexpr int H_  = 64;
constexpr int W_  = 64;
constexpr int K_  = 1024;
constexpr int HW_ = H_ * W_;          // 4096 = 2^12
constexpr int N_  = B_ * HW_;         // 32768 pixels
constexpr int C2_ = 2 * C_;           // 512

typedef _Float16 f16;
typedef __attribute__((ext_vector_type(16))) _Float16 v16h;
typedef __attribute__((ext_vector_type(8)))  _Float16 v8h;
typedef __attribute__((ext_vector_type(8)))  float    v8f;

// ---------------------------------------------------------------------------
// Register-blocked WMMA row-block: one 128-thread block computes a 16 x 256
// output slab. The 16-row A tile (16 x KD f16) is shared by all 4 waves, so it
// is staged ONCE into LDS with gfx1250 async global->LDS DMA (ASYNCcnt), then
// each wave computes 4 adjacent 16x16 tiles (4 accumulators) reusing the same
// LDS A fragment per k-step: 4 WMMAs per 128B of global B traffic.
//
// A  : row-major (M x lda)
// Bt : stored (out_dim x K) row-major, so B column j == contiguous row j.
// ---------------------------------------------------------------------------
template <int KD>
__device__ __forceinline__ void wmma_rowblock(const f16* __restrict__ A, int lda,
                                              int row0,
                                              const f16* __restrict__ Bt, int ldb,
                                              int colbase, f16* shA, v8f acc[4]) {
  const int tid  = threadIdx.x;     // 128 threads = 4 waves
  const int lane = tid & 31;
  const int wave = tid >> 5;

  // ---- stage A tile (16 x KD) global -> LDS via async DMA ----
  constexpr int CHUNKS = 16 * KD / 8;            // 16-byte chunks
  for (int q = tid; q < CHUNKS; q += 128) {
    const int row = q / (KD / 8);
    const int ck  = q % (KD / 8);
    const f16* gsrc = A + (size_t)(row0 + row) * lda + ck * 8;
    // low 32 bits of a flat LDS pointer == hardware LDS byte offset
    const unsigned ldsoff = (unsigned)(size_t)(shA + row * KD + ck * 8);
    asm volatile("global_load_async_to_lds_b128 %0, %1, off"
                 :: "v"(ldsoff), "v"((unsigned long long)(size_t)gsrc)
                 : "memory");
  }
  asm volatile("s_wait_asynccnt 0x0" ::: "memory");
  __syncthreads();

  // ---- per-lane fragment base pointers ----
  // 16-bit A 16x32 layout: lanes 0-15 hold K[0..8)+K[16..24), lanes 16-31 rest
  const int ka = (lane < 16) ? 0 : 8;
  // 16-bit B 32x16 layout: lanes 0-15 hold K[0..8)+K[16..24), lanes 16-31 rest
  const int kb = (lane < 16) ? 0 : 8;
  const f16* a0 = shA + (lane & 15) * KD + ka;   // ds_load_b128 fragments
  const f16* bp0 = Bt + (size_t)(colbase + wave * 64 +  0 + (lane & 15)) * ldb + kb;
  const f16* bp1 = Bt + (size_t)(colbase + wave * 64 + 16 + (lane & 15)) * ldb + kb;
  const f16* bp2 = Bt + (size_t)(colbase + wave * 64 + 32 + (lane & 15)) * ldb + kb;
  const f16* bp3 = Bt + (size_t)(colbase + wave * 64 + 48 + (lane & 15)) * ldb + kb;

  for (int k0 = 0; k0 < KD; k0 += 32) {
    __builtin_prefetch(bp0 + k0 + 32, 0, 1);     // global_prefetch_b8
    __builtin_prefetch(bp2 + k0 + 32, 0, 1);
    v8h alo = *(const v8h*)(a0 + k0);
    v8h ahi = *(const v8h*)(a0 + k0 + 16);
    v16h a = __builtin_shufflevector(alo, ahi, 0,1,2,3,4,5,6,7,8,9,10,11,12,13,14,15);
    const f16* bps[4] = {bp0, bp1, bp2, bp3};
#pragma unroll
    for (int t = 0; t < 4; ++t) {
      v8h blo = *(const v8h*)(bps[t] + k0);
      v8h bhi = *(const v8h*)(bps[t] + k0 + 16);
      v16h b = __builtin_shufflevector(blo, bhi, 0,1,2,3,4,5,6,7,8,9,10,11,12,13,14,15);
      acc[t] = __builtin_amdgcn_wmma_f32_16x16x32_f16(false, a, false, b,
                                                      (short)0, acc[t], false, false);
    }
  }
}

// ---------------------------------------------------------------------------
// 1) transpose x (B,C,H,W) -> xn (n,C) f16 row-normalized, + row norms
// ---------------------------------------------------------------------------
__global__ void k_xnorm(const float* __restrict__ x, f16* __restrict__ xn,
                        float* __restrict__ rn) {
  __shared__ float red[256];
  __shared__ float inv;
  const int n = blockIdx.x, c = threadIdx.x;
  const int b = n >> 12, hw = n & 4095;
  const float v = x[((size_t)(b * C_ + c) << 12) + hw];
  red[c] = v * v;
  __syncthreads();
  for (int s = 128; s; s >>= 1) { if (c < s) red[c] += red[c + s]; __syncthreads(); }
  if (c == 0) { float nr = fmaxf(sqrtf(red[0]), 1e-12f); rn[n] = nr; inv = 1.0f / nr; }
  __syncthreads();
  xn[(size_t)n * C_ + c] = (f16)(v * inv);
}

// 2) mn0 = l2norm(units_w) rows -> f16
__global__ void k_munit(const float* __restrict__ uw, f16* __restrict__ mn0) {
  __shared__ float red[256];
  __shared__ float inv;
  const int k = blockIdx.x, c = threadIdx.x;
  const float v = uw[(size_t)k * C_ + c];
  red[c] = v * v;
  __syncthreads();
  for (int s = 128; s; s >>= 1) { if (c < s) red[c] += red[c + s]; __syncthreads(); }
  if (c == 0) inv = 1.0f / fmaxf(sqrtf(red[0]), 1e-12f);
  __syncthreads();
  mn0[(size_t)k * C_ + c] = (f16)(v * inv);
}

__global__ void k_cvt_f16(const float* __restrict__ s, f16* __restrict__ d, int n) {
  int i = blockIdx.x * 256 + threadIdx.x;
  if (i < n) d[i] = (f16)s[i];
}

// ---------------------------------------------------------------------------
// 3) score = xn @ mn0^T   (n x 1024), written directly into output slot 2
// ---------------------------------------------------------------------------
__global__ void k_gemm_score(const f16* __restrict__ xn, const f16* __restrict__ mn0,
                             float* __restrict__ score) {
  __shared__ f16 shA[16 * C_];
  const int colbase = blockIdx.x * 256;
  const int row0 = blockIdx.y * 16;
  v8f acc[4] = {};
  wmma_rowblock<C_>(xn, C_, row0, mn0, C_, colbase, shA, acc);
  const int lane = threadIdx.x & 31, wave = threadIdx.x >> 5;
  const int r0 = row0 + ((lane >> 4) << 3);
#pragma unroll
  for (int t = 0; t < 4; ++t) {
    const int cc = colbase + wave * 64 + t * 16 + (lane & 15);
    for (int r = 0; r < 8; ++r)
      score[(size_t)(r0 + r) * K_ + cc] = acc[t][r];
  }
}

// 4) g = (xn @ g_w^T)*||x||_row + g_b, scattered into BCHW layout
__global__ void k_gemm_g(const f16* __restrict__ xn, const f16* __restrict__ gw,
                         const float* __restrict__ rn, const float* __restrict__ gb,
                         float* __restrict__ gbchw) {
  __shared__ f16 shA[16 * C_];
  const int row0 = blockIdx.y * 16;
  v8f acc[4] = {};
  wmma_rowblock<C_>(xn, C_, row0, gw, C_, 0, shA, acc);
  const int lane = threadIdx.x & 31, wave = threadIdx.x >> 5;
  const int r0 = row0 + ((lane >> 4) << 3);
#pragma unroll
  for (int t = 0; t < 4; ++t) {
    const int cc = wave * 64 + t * 16 + (lane & 15);
    const float bias = gb[cc];
    for (int r = 0; r < 8; ++r) {
      const int row = r0 + r;
      const int b = row >> 12, hw = row & 4095;
      gbchw[((size_t)(b * C_ + cc) << 12) + hw] = acc[t][r] * rn[row] + bias;
    }
  }
}

// ---------------------------------------------------------------------------
// 5) wave-per-row argmax over K=1024 (first-index tie-break like jnp.argmax)
// ---------------------------------------------------------------------------
__global__ void k_argmax(const float* __restrict__ score, int* __restrict__ ind,
                         float* __restrict__ counts) {
  const int row  = (blockIdx.x * blockDim.x + threadIdx.x) >> 5;
  const int lane = threadIdx.x & 31;
  const float* sr = score + (size_t)row * K_;
  float best = -3.4e38f; int bidx = 0;
  for (int k = lane; k < K_; k += 32) {
    float v = sr[k];
    if (v > best) { best = v; bidx = k; }
  }
  for (int off = 16; off; off >>= 1) {
    float ob = __shfl_down(best, off, 32);
    int   oi = __shfl_down(bidx, off, 32);
    if (ob > best || (ob == best && oi < bidx)) { best = ob; bidx = oi; }
  }
  if (lane == 0) { ind[row] = bidx; atomicAdd(&counts[bidx], 1.0f); }
}

// 6) segment sums of detached xf into codebook bins
__global__ void k_scatter(const f16* __restrict__ xn, const float* __restrict__ rn,
                          const int* __restrict__ ind, float* __restrict__ sums) {
  const int n = blockIdx.x, c = threadIdx.x;
  const float v = (float)xn[(size_t)n * C_ + c] * rn[n];
  atomicAdd(&sums[(size_t)ind[n] * C_ + c], v);
}

// ---------------------------------------------------------------------------
// 7) EMA update m, renormalize, build kk = [relu(mn)^p, relu(-mn)^p], kmean
// ---------------------------------------------------------------------------
__global__ void k_update(const float* __restrict__ uw, const float* __restrict__ sums,
                         const float* __restrict__ counts, const float* __restrict__ power,
                         float* __restrict__ mnew, float* __restrict__ kk,
                         float* __restrict__ kmean) {
  __shared__ float red[256];
  __shared__ float inv;
  const int k = blockIdx.x, c = threadIdx.x;
  const float em = sums[(size_t)k * C_ + c] / (counts[k] + 1e-6f);
  const float m = uw[(size_t)k * C_ + c] * 0.999f + em * 0.001f;
  mnew[(size_t)k * C_ + c] = m;
  red[c] = m * m;
  __syncthreads();
  for (int s = 128; s; s >>= 1) { if (c < s) red[c] += red[c + s]; __syncthreads(); }
  if (c == 0) inv = 1.0f / fmaxf(sqrtf(red[0]), 1e-12f);
  __syncthreads();
  const float mn = m * inv;
  const float p  = 1.0f + 4.0f / (1.0f + __expf(-power[c]));
  const float kp = (mn > 0.0f) ? __powf(mn, p)  : 0.0f;
  const float kn = (mn < 0.0f) ? __powf(-mn, p) : 0.0f;
  kk[(size_t)k * C2_ + c]       = kp;
  kk[(size_t)k * C2_ + C_ + c]  = kn;
  atomicAdd(&kmean[c],       kp * (1.0f / (float)K_));
  atomicAdd(&kmean[C_ + c],  kn * (1.0f / (float)K_));
}

// ---------------------------------------------------------------------------
// 8) kvT (256 x 512) f16: cols<128 -> kv (v1), cols>=128 -> row-swapped kv2 (v2)
//    (q_opp half-swap folded into B so x_sim/x_opp become ONE WMMA GEMM)
// ---------------------------------------------------------------------------
__global__ void k_build_kv(const float* __restrict__ kk, const float* __restrict__ mnew,
                           f16* __restrict__ kvT) {
  const int j = blockIdx.x, c2 = threadIdx.x;                 // j in [0,512), c2 in [0,256)
  const int jj = (c2 < 128) ? j : ((j + 256) & 511);
  float acc = 0.0f;
  for (int k = 0; k < K_; ++k)
    acc += kk[(size_t)k * C2_ + jj] * mnew[(size_t)k * C_ + c2];
  kvT[(size_t)c2 * C2_ + j] = (f16)(acc * (1.0f / (float)N_));  // s^2 = 1/n
}

// ---------------------------------------------------------------------------
// 9) q features (f16) + fused z / z2 normalizers (dot with kmean, half-swapped)
// ---------------------------------------------------------------------------
__global__ void k_q(const f16* __restrict__ xn, const float* __restrict__ power,
                    const float* __restrict__ kmean, f16* __restrict__ qsim,
                    float* __restrict__ zbuf) {
  __shared__ float rz[256];
  __shared__ float rz2[256];
  const int n = blockIdx.x, c = threadIdx.x;
  const float xv = (float)xn[(size_t)n * C_ + c];
  const float p  = 1.0f + 4.0f / (1.0f + __expf(-power[c]));
  const float qp = (xv > 0.0f) ? __powf(xv, p)  : 0.0f;
  const float qn = (xv < 0.0f) ? __powf(-xv, p) : 0.0f;
  qsim[(size_t)n * C2_ + c]      = (f16)qp;
  qsim[(size_t)n * C2_ + C_ + c] = (f16)qn;
  const float km0 = kmean[c], km1 = kmean[C_ + c];
  rz[c]  = qp * km0 + qn * km1;   // q_sim . kmean
  rz2[c] = qp * km1 + qn * km0;   // q_opp . kmean
  __syncthreads();
  for (int s = 128; s; s >>= 1) {
    if (c < s) { rz[c] += rz[c + s]; rz2[c] += rz2[c + s]; }
    __syncthreads();
  }
  if (c == 0) {
    zbuf[2 * n]     = 1.0f / (rz[0]  + 1e-6f);
    zbuf[2 * n + 1] = 1.0f / (rz2[0] + 1e-6f);
  }
}

// 10) xo = (qsim @ kvcat) * z/z2, scattered into BCHW layout
__global__ void k_gemm_xo(const f16* __restrict__ qsim, const f16* __restrict__ kvT,
                          const float* __restrict__ zbuf, float* __restrict__ xo) {
  __shared__ f16 shA[16 * C2_];
  const int row0 = blockIdx.y * 16;
  v8f acc[4] = {};
  wmma_rowblock<C2_>(qsim, C2_, row0, kvT, C2_, 0, shA, acc);
  const int lane = threadIdx.x & 31, wave = threadIdx.x >> 5;
  const int r0 = row0 + ((lane >> 4) << 3);
#pragma unroll
  for (int t = 0; t < 4; ++t) {
    const int cc = wave * 64 + t * 16 + (lane & 15);
    const int zsel = (cc < 128) ? 0 : 1;
    for (int r = 0; r < 8; ++r) {
      const int row = r0 + r;
      const int b = row >> 12, hw = row & 4095;
      xo[((size_t)(b * C_ + cc) << 12) + hw] = acc[t][r] * zbuf[2 * row + zsel];
    }
  }
}

// 11) depthwise 3x3 (SAME, cross-correlation) + residual + gate
__global__ void k_final(const float* __restrict__ xo, const float* __restrict__ g,
                        const float* __restrict__ dwcw, const float* __restrict__ dwcb,
                        float* __restrict__ out) {
  const size_t i = (size_t)blockIdx.x * 256 + threadIdx.x;
  const int w = i & 63, h = (int)((i >> 6) & 63), c = (int)((i >> 12) & 255);
  const int b = (int)(i >> 20);
  const float* plane = xo + ((size_t)(b * C_ + c) << 12);
  float acc = dwcb[c];
  for (int dy = -1; dy <= 1; ++dy)
    for (int dx = -1; dx <= 1; ++dx) {
      const int hh = h + dy, ww = w + dx;
      if (hh >= 0 && hh < 64 && ww >= 0 && ww < 64)
        acc += plane[hh * 64 + ww] * dwcw[c * 9 + (dy + 1) * 3 + (dx + 1)];
    }
  const float v = plane[h * 64 + w];
  out[i] = (v + acc) * g[i];
}

// ---------------------------------------------------------------------------
extern "C" void kernel_launch(void* const* d_in, const int* in_sizes, int n_in,
                              void* d_out, int out_size, void* d_ws, size_t ws_size,
                              hipStream_t stream) {
  (void)in_sizes; (void)n_in; (void)out_size;
  const float* x     = (const float*)d_in[0];
  const float* uw    = (const float*)d_in[1];
  const float* power = (const float*)d_in[2];
  const float* g_w   = (const float*)d_in[3];
  const float* g_b   = (const float*)d_in[4];
  const float* dwcw  = (const float*)d_in[5];
  const float* dwcb  = (const float*)d_in[6];

  float* out   = (float*)d_out;
  float* score = out + (size_t)B_ * C_ * HW_;   // second returned tensor

  char* ws = (char*)d_ws;
  size_t off = 0;
  auto take = [&](size_t bytes) -> char* {
    char* p = ws + off;
    off = (off + bytes + 255) & ~(size_t)255;
    return p;
  };
  f16*   xn_h   = (f16*)  take((size_t)N_ * C_ * 2);
  float* rn     = (float*)take((size_t)N_ * 4);
  f16*   mn0_h  = (f16*)  take((size_t)K_ * C_ * 2);
  f16*   gw_h   = (f16*)  take((size_t)C_ * C_ * 2);
  int*   ind    = (int*)  take((size_t)N_ * 4);
  float* counts = (float*)take((size_t)K_ * 4);
  float* sums   = (float*)take((size_t)K_ * C_ * 4);
  float* mnew   = (float*)take((size_t)K_ * C_ * 4);
  float* kkbuf  = (float*)take((size_t)K_ * C2_ * 4);
  float* kmean  = (float*)take((size_t)C2_ * 4);
  f16*   kvT    = (f16*)  take((size_t)C_ * C2_ * 2);
  f16*   qsim   = (f16*)  take((size_t)N_ * C2_ * 2);
  float* zbuf   = (float*)take((size_t)N_ * 2 * 4);
  float* xo     = (float*)take((size_t)N_ * C_ * 4);
  float* gbchw  = (float*)take((size_t)N_ * C_ * 4);
  if (off > ws_size) return;   // workspace too small: bail deterministically

  hipMemsetAsync(counts, 0, (size_t)K_ * 4, stream);
  hipMemsetAsync(sums,   0, (size_t)K_ * C_ * 4, stream);
  hipMemsetAsync(kmean,  0, (size_t)C2_ * 4, stream);

  k_xnorm  <<<N_, 256, 0, stream>>>(x, xn_h, rn);
  k_munit  <<<K_, 256, 0, stream>>>(uw, mn0_h);
  k_cvt_f16<<<(C_ * C_) / 256, 256, 0, stream>>>(g_w, gw_h, C_ * C_);

  // 128-thread blocks: 4 waves x (16x64) = 16 x 256 output slab per block
  k_gemm_score<<<dim3(K_ / 256, N_ / 16), 128, 0, stream>>>(xn_h, mn0_h, score);
  k_gemm_g    <<<dim3(1,        N_ / 16), 128, 0, stream>>>(xn_h, gw_h, rn, g_b, gbchw);

  k_argmax <<<N_ / 8, 256, 0, stream>>>(score, ind, counts);
  k_scatter<<<N_,     256, 0, stream>>>(xn_h, rn, ind, sums);
  k_update <<<K_,     256, 0, stream>>>(uw, sums, counts, power, mnew, kkbuf, kmean);

  k_q       <<<N_, 256, 0, stream>>>(xn_h, power, kmean, qsim, zbuf);
  k_build_kv<<<C2_, 256, 0, stream>>>(kkbuf, mnew, kvT);

  k_gemm_xo<<<dim3(1, N_ / 16), 128, 0, stream>>>(qsim, kvT, zbuf, xo);
  k_final  <<<(B_ * C_ * HW_) / 256, 256, 0, stream>>>(xo, gbchw, dwcw, dwcb, out);
}
